// OptimizedCDCConv2d_17918603559476
// MI455X (gfx1250) — matrix operand order repaired and checked by reading, not compiled, not run on hardware
//
#include <hip/hip_runtime.h>
#include <math.h>

// ---------------- problem constants ----------------
#define BB    8
#define CIN   64
#define COUT  64
#define HH    256
#define WW    256
#define CRED  8
#define THETA 0.7f
#define TP    130           // LDS pixels per row: 128 + 2 halo

// ---------------- WMMA types ----------------
typedef __attribute__((ext_vector_type(16))) __bf16        v16bf;
typedef __attribute__((ext_vector_type(8)))  float         v8f;
typedef __attribute__((ext_vector_type(8)))  unsigned int  v8u;

union Frag { v8u u; v16bf b; };

__device__ __forceinline__ unsigned short f2bf(float f) {
    unsigned u = __float_as_uint(f);
    unsigned r = (u + 0x7FFFu + ((u >> 16) & 1u)) >> 16;   // RNE
    return (unsigned short)r;
}
__device__ __forceinline__ float bflo(unsigned u) { return __uint_as_float(u << 16); }
__device__ __forceinline__ float bfhi(unsigned u) { return __uint_as_float(u & 0xFFFF0000u); }

// ---------------------------------------------------------------------------
// Prep: swizzle theta*W_conv and (1-theta)*W_edge into WMMA A-fragment order.
// A 16x32 bf16 layout: lane = M (mod 16); K = (lane>=16?16:0)+vgpr*2+half
// Stored so lane l reads its 16 bf16 contiguously at [frag*512 + l*16].
// ---------------------------------------------------------------------------
__global__ void cdc_prep(const float* __restrict__ Wc_in,
                         const float* __restrict__ We_in,
                         unsigned short* __restrict__ Wc,
                         unsigned short* __restrict__ We) {
    int g = blockIdx.x * 256 + threadIdx.x;
    if (g < 36864) {                         // 9 offsets * 2 kc * 4 mt * 512
        int i    = g & 15;
        int lane = (g >> 4) & 31;
        int mt   = (g >> 9) & 3;
        int kc   = (g >> 11) & 1;
        int off  = g >> 12;                  // 0..8
        int ky = off / 3, kx = off % 3;
        int m  = mt * 16 + (lane & 15);
        int vg = i >> 1, hf = i & 1;
        int k  = ((lane & 16) ? 16 : 0) + (vg << 1) + hf;
        int c  = kc * 32 + k;
        Wc[g] = f2bf(THETA * Wc_in[((m * CIN + c) * 3 + ky) * 3 + kx]);
    } else if (g < 36864 + 4096) {           // edge: 2 kc * 4 mt * 512
        int e    = g - 36864;
        int i    = e & 15;
        int lane = (e >> 4) & 31;
        int mt   = (e >> 9) & 3;
        int kc   = (e >> 11) & 1;
        int m  = mt * 16 + (lane & 15);
        int vg = i >> 1, hf = i & 1;
        int k  = ((lane & 16) ? 16 : 0) + (vg << 1) + hf;
        int c  = kc * 32 + k;
        We[e] = f2bf((1.0f - THETA) * We_in[m * CIN + c]);
    }
}

// ---------------------------------------------------------------------------
// Main: per block (b, y, half-row of 128 px). 8 waves; each wave owns 16 px
// and all 64 output channels (4 M-tiles). combined -> d_out, deterministic
// per-block channel sums -> blocksums.
// ---------------------------------------------------------------------------
__global__ __launch_bounds__(256) void cdc_main(const float* __restrict__ x,
                                                const float* __restrict__ b_conv,
                                                const unsigned short* __restrict__ Wc,
                                                const unsigned short* __restrict__ We,
                                                float* __restrict__ out,
                                                float* __restrict__ blocksums) {
    __shared__ unsigned short sx[3 * TP * CIN];   // [row][pixel][channel] bf16
    __shared__ float wsum[8][COUT];

    const int bid = blockIdx.x;          // b*512 + y*2 + xh
    const int xh  = bid & 1;
    const int y   = (bid >> 1) & 255;
    const int b   = bid >> 9;
    const int x0  = xh * 128;
    const int t   = threadIdx.x;

    // ---- stage x tile (3 rows, halo, f32 -> bf16, transpose to pixel-major)
    for (int e = t; e < 3 * CIN * TP; e += 256) {
        int px = e % TP;
        int c  = (e / TP) % CIN;
        int r  = e / (TP * CIN);
        int gx = x0 - 1 + px;
        int gy = y - 1 + r;
        float v = 0.0f;
        if ((unsigned)gx < WW && (unsigned)gy < HH)
            v = x[(((b * CIN + c) * HH + gy) << 8) + gx];
        sx[(r * TP + px) * CIN + c] = f2bf(v);
    }
    __syncthreads();

    const int wave = t >> 5;
    const int lane = t & 31;
    const int p0   = wave * 16;          // wave pixel base in [0,128)
    const int n    = lane & 15;          // N index (pixel within tile)
    const int kb   = lane & 16;          // channel sub-offset for B frag

    v8f acc[4];
    #pragma unroll
    for (int mt = 0; mt < 4; ++mt) {
        v8f z = {0.f,0.f,0.f,0.f,0.f,0.f,0.f,0.f};
        acc[mt] = z;
    }

    // ---- 3x3 conv as 9 GEMMs (theta folded into weights)
    #pragma unroll
    for (int ky = 0; ky < 3; ++ky) {
        #pragma unroll
        for (int kx = 0; kx < 3; ++kx) {
            const int p = p0 + n + kx;   // LDS pixel index (halo offset included)
            #pragma unroll
            for (int kc = 0; kc < 2; ++kc) {
                const int cb = kc * 32 + kb;
                Frag bf;
                bf.u = *(const v8u*)&sx[(ky * TP + p) * CIN + cb];
                const unsigned short* wb =
                    Wc + (((ky * 3 + kx) * 2 + kc) * 4) * 512 + lane * 16;
                #pragma unroll
                for (int mt = 0; mt < 4; ++mt) {
                    Frag af;
                    af.u = *(const v8u*)(wb + mt * 512);
                    acc[mt] = __builtin_amdgcn_wmma_f32_16x16x32_bf16(
                        false, af.b, false, bf.b, (short)0, acc[mt], false, false);
                }
            }
        }
    }

    // ---- edge term: |x(y-1) - x(y+1)|, 1x1 conv ((1-theta) folded in)
    {
        const int pc = p0 + n + 1;       // center pixel in LDS coords
        #pragma unroll
        for (int kc = 0; kc < 2; ++kc) {
            const int cb = kc * 32 + kb;
            v8u r0 = *(const v8u*)&sx[(0 * TP + pc) * CIN + cb];
            v8u r2 = *(const v8u*)&sx[(2 * TP + pc) * CIN + cb];
            Frag bf;
            #pragma unroll
            for (int i = 0; i < 8; ++i) {
                unsigned a = r0[i], c2 = r2[i];
                unsigned short dlo = f2bf(fabsf(bflo(a) - bflo(c2)));
                unsigned short dhi = f2bf(fabsf(bfhi(a) - bfhi(c2)));
                bf.u[i] = (unsigned)dlo | ((unsigned)dhi << 16);
            }
            const unsigned short* wb = We + (kc * 4) * 512 + lane * 16;
            #pragma unroll
            for (int mt = 0; mt < 4; ++mt) {
                Frag af;
                af.u = *(const v8u*)(wb + mt * 512);
                acc[mt] = __builtin_amdgcn_wmma_f32_16x16x32_bf16(
                    false, af.b, false, bf.b, (short)0, acc[mt], false, false);
            }
        }
    }

    // ---- bias, store combined, deterministic channel sums
    const int gx = x0 + p0 + n;
    const int mhalf = kb ? 8 : 0;
    #pragma unroll
    for (int mt = 0; mt < 4; ++mt) {
        #pragma unroll
        for (int j = 0; j < 8; ++j) {
            const int m = mt * 16 + j + mhalf;
            float v = acc[mt][j] + THETA * b_conv[m];
            out[(((b * COUT + m) * HH + y) << 8) + gx] = v;
            // reduce over the 16 lanes of this half (xor 1/2/4/8 stays in half)
            float s = v;
            s += __shfl_xor(s, 1, 32);
            s += __shfl_xor(s, 2, 32);
            s += __shfl_xor(s, 4, 32);
            s += __shfl_xor(s, 8, 32);
            if (n == 0) wsum[wave][m] = s;
        }
    }
    __syncthreads();
    if (t < COUT) {
        float s = 0.0f;
        #pragma unroll
        for (int w = 0; w < 8; ++w) s += wsum[w][t];
        blocksums[bid * COUT + t] = s;
    }
}

// ---------------------------------------------------------------------------
// Deterministic pooled reduction: pooled[b][m] = sum over 512 blocks of b.
// ---------------------------------------------------------------------------
__global__ void cdc_pool(const float* __restrict__ blocksums,
                         float* __restrict__ pooled) {
    __shared__ float red[128];
    const int bm = blockIdx.x;           // b*64 + m
    const int b  = bm >> 6, m = bm & 63;
    float s = 0.0f;
    for (int i = threadIdx.x; i < 512; i += 128)
        s += blocksums[(b * 512 + i) * COUT + m];
    red[threadIdx.x] = s;
    __syncthreads();
    for (int off = 64; off > 0; off >>= 1) {
        if (threadIdx.x < off) red[threadIdx.x] += red[threadIdx.x + off];
        __syncthreads();
    }
    if (threadIdx.x == 0) pooled[bm] = red[0];
}

// ---------------------------------------------------------------------------
// SE attention: mean -> FC(64->8) ReLU -> FC(8->64) sigmoid.
// ---------------------------------------------------------------------------
__global__ void cdc_attn(const float* __restrict__ pooled,
                         const float* __restrict__ W1, const float* __restrict__ b1,
                         const float* __restrict__ W2, const float* __restrict__ b2,
                         float* __restrict__ attn) {
    __shared__ float hsh[BB * CRED];
    const int t = threadIdx.x;           // 512 threads
    if (t < BB * CRED) {
        int b = t >> 3, r = t & 7;
        float s = b1[r];
        const float inv = 1.0f / (float)(HH * WW);
        for (int c = 0; c < COUT; ++c)
            s += pooled[b * COUT + c] * inv * W1[r * COUT + c];
        hsh[t] = s > 0.0f ? s : 0.0f;
    }
    __syncthreads();
    int b = t >> 6, c = t & 63;
    float s = b2[c];
    #pragma unroll
    for (int r = 0; r < CRED; ++r) s += hsh[b * CRED + r] * W2[c * CRED + r];
    attn[t] = 1.0f / (1.0f + expf(-s));
}

// ---------------------------------------------------------------------------
// In-place rescale of combined by attn[b][c] (float4 vectorized).
// ---------------------------------------------------------------------------
__global__ void cdc_scale(float* __restrict__ out, const float* __restrict__ attn) {
    const long long i4 = (long long)blockIdx.x * 256 + threadIdx.x; // 8388608 total
    const int chan = (int)(i4 >> 14);     // (i4*4)/65536 = b*64+c
    const float a = attn[chan];
    float4 v = ((float4*)out)[i4];
    v.x *= a; v.y *= a; v.z *= a; v.w *= a;
    ((float4*)out)[i4] = v;
}

// ---------------------------------------------------------------------------
extern "C" void kernel_launch(void* const* d_in, const int* in_sizes, int n_in,
                              void* d_out, int out_size, void* d_ws, size_t ws_size,
                              hipStream_t stream) {
    const float* x     = (const float*)d_in[0];
    const float* Wconv = (const float*)d_in[1];
    const float* bconv = (const float*)d_in[2];
    const float* Wedge = (const float*)d_in[3];
    const float* W1    = (const float*)d_in[4];
    const float* b1    = (const float*)d_in[5];
    const float* W2    = (const float*)d_in[6];
    const float* b2    = (const float*)d_in[7];
    float* out = (float*)d_out;

    char* ws = (char*)d_ws;
    unsigned short* Wc     = (unsigned short*)(ws);            // 73728 B
    unsigned short* We     = (unsigned short*)(ws + 73728);    //  8192 B
    float*          pooled = (float*)(ws + 81920);             //  2048 B
    float*          attn   = (float*)(ws + 83968);             //  2048 B
    float*          bsums  = (float*)(ws + 86016);             // 4096*64*4 = 1 MB

    cdc_prep <<<160,   256, 0, stream>>>(Wconv, Wedge, Wc, We);
    cdc_main <<<4096,  256, 0, stream>>>(x, bconv, Wc, We, out, bsums);
    cdc_pool <<<512,   128, 0, stream>>>(bsums, pooled);
    cdc_attn <<<1,     512, 0, stream>>>(pooled, W1, b1, W2, b2, attn);
    cdc_scale<<<32768, 256, 0, stream>>>(out, attn);
}
